// PatchWeighted_76166950027380
// MI455X (gfx1250) — compile-verified, matrix operand order checked by reading, not run on hardware
//
#include <hip/hip_runtime.h>
#include <cstdint>
#include <cstddef>

// ---------------------------------------------------------------------------
// Types for CDNA5 WMMA (wave32): v_wmma_f32_16x16x32_bf16
// ---------------------------------------------------------------------------
typedef __attribute__((ext_vector_type(16))) __bf16 bf16x16;
typedef __attribute__((ext_vector_type(8)))  __bf16 bf16x8;
typedef __attribute__((ext_vector_type(8)))  float  f32x8;

#define DEV __device__ __forceinline__

// ---------------------------------------------------------------------------
// Problem constants
// ---------------------------------------------------------------------------
#define BATCH   2
#define CDIM    768
#define DDIM    16
#define HDIM    14
#define WDIM    14
#define TOKENS  6272        // B*D*H*W
#define NWIN    16          // total windows (2 batches * 8)
#define NTOK    392         // 8*7*7 tokens per window
#define NPAD    400         // NTOK padded to mult of 16
#define KPAD    416         // NTOK padded to mult of 32 (AV K dim)
#define NHEADS  24
#define HD      32          // head dim == WMMA K for bf16. exact fit.
#define MWIN    6400        // NWIN*NPAD
#define QKVN    2304
#define MLPN    3072
#define QSCALE  0.17677669529663687f   // 32^-0.5

// ---------------------------------------------------------------------------
// Device helpers
// ---------------------------------------------------------------------------
union FragU { bf16x16 v; bf16x8 h[2]; };

// Load a 16x32 bf16 WMMA fragment (A layout; identical pattern for B when the
// weight is row-major N x K, i.e. computing X @ W^T). Per ISA 16-bit A table:
// lane<16: row=lane,    K = [0..7] U [16..23]
// lane>=16: row=lane-16, K = [8..15] U [24..31]
DEV bf16x16 load_frag(const __bf16* base, int stride, int lane) {
  const __bf16* p = base + (size_t)(lane & 15) * stride + ((lane >> 4) << 3);
  FragU u;
  u.h[0] = *reinterpret_cast<const bf16x8*>(p);
  u.h[1] = *reinterpret_cast<const bf16x8*>(p + 16);
  return u.v;
}

DEV float waveSum(float s) {
  #pragma unroll
  for (int o = 16; o; o >>= 1) s += __shfl_xor(s, o);
  return s;
}

DEV float geluf(float x) { return 0.5f * x * (1.0f + erff(x * 0.70710678118654752f)); }

// window token -> spatial token index (with cyclic shift for shifted blocks)
DEV int mapToken(int win, int t, int shifted) {
  int b  = win >> 3;
  int wr = win & 7;
  int wd = (wr >> 2) & 1, wh = (wr >> 1) & 1, ww = wr & 1;
  int id = t / 49, rem = t % 49, ih = rem / 7, iw = rem % 7;
  int dd = wd * 8 + id, hh = wh * 7 + ih, wx = ww * 7 + iw;
  if (shifted) { dd = (dd + 4) & 15; hh = (hh + 3) % 14; wx = (wx + 3) % 14; }
  return ((b * DDIM + dd) * HDIM + hh) * WDIM + wx;
}

// Swin shift-mask region label (computed in rolled coordinates)
DEV int labelOf(int win, int t) {
  int wr = win & 7;
  int wd = (wr >> 2) & 1, wh = (wr >> 1) & 1, ww = wr & 1;
  int id = t / 49, rem = t % 49, ih = rem / 7, iw = rem % 7;
  int dd = wd * 8 + id, hh = wh * 7 + ih, wx = ww * 7 + iw;
  int ld = dd < 8 ? 0 : (dd < 12 ? 1 : 2);
  int lh = hh < 7 ? 0 : (hh < 11 ? 1 : 2);
  int lw = wx < 7 ? 0 : (wx < 11 ? 1 : 2);
  return 9 * ld + 3 * lh + lw;
}

// ---------------------------------------------------------------------------
// Elementwise kernels
// ---------------------------------------------------------------------------
__global__ __launch_bounds__(256) void k_cast_bf16(const float* __restrict__ s,
                                                   __bf16* __restrict__ d, int n) {
  int i = blockIdx.x * 256 + threadIdx.x;
  if (i < n) d[i] = (__bf16)s[i];
}

// x (B,C,D,H,W) f32 -> bufX token-major (B,D,H,W,C) f32
__global__ __launch_bounds__(256) void k_init_x(const float* __restrict__ x,
                                                float* __restrict__ bufX) {
  int o = blockIdx.x * 256 + threadIdx.x;
  if (o >= TOKENS * CDIM) return;
  int tok = o / CDIM, c = o % CDIM;
  int b = tok / (DDIM * HDIM * WDIM);
  int r = tok % (DDIM * HDIM * WDIM);
  bufX[o] = x[((size_t)(b * CDIM + c)) * (DDIM * HDIM * WDIM) + r];
}

// LayerNorm (+optional shift/window partition) -> bf16. One wave per token row.
// mode 0: identity over TOKENS rows. mode 1/2: window partition (un)shifted, MWIN rows.
__global__ __launch_bounds__(256) void k_ln(const float* __restrict__ src,
                                            const float* __restrict__ g,
                                            const float* __restrict__ b,
                                            __bf16* __restrict__ dst,
                                            int mode, int Mtok) {
  const int lane = threadIdx.x & 31;
  const int wv = threadIdx.x >> 5;
  const int tk = blockIdx.x * 8 + wv;
  if (tk >= Mtok) return;
  __bf16* o = dst + (size_t)tk * CDIM;
  int tok;
  if (mode == 0) tok = tk;
  else {
    int win = tk / NPAD, t = tk % NPAD;
    if (t >= NTOK) {
      #pragma unroll
      for (int i = 0; i < 24; ++i) o[lane + 32 * i] = (__bf16)0.0f;
      return;
    }
    tok = mapToken(win, t, mode == 2);
  }
  const float* x = src + (size_t)tok * CDIM;
  float v[24]; float s = 0.f;
  #pragma unroll
  for (int i = 0; i < 24; ++i) { v[i] = x[lane + 32 * i]; s += v[i]; }
  float mu = waveSum(s) * (1.0f / CDIM);
  float q = 0.f;
  #pragma unroll
  for (int i = 0; i < 24; ++i) { float d = v[i] - mu; q += d * d; }
  float rs = rsqrtf(waveSum(q) * (1.0f / CDIM) + 1e-5f);
  #pragma unroll
  for (int i = 0; i < 24; ++i) {
    int c = lane + 32 * i;
    o[c] = (__bf16)((v[i] - mu) * rs * g[c] + b[c]);
  }
}

// Vt[(win,head,d,j)] = V[(win,j,head,d)], zero-padded keys j>=392
__global__ __launch_bounds__(256) void k_vt(const __bf16* __restrict__ qkv,
                                            __bf16* __restrict__ Vt) {
  int idx = blockIdx.x * 256 + threadIdx.x;
  if (idx >= NWIN * NHEADS * HD * KPAD) return;
  int j = idx % KPAD; int rest = idx / KPAD;
  int d = rest % HD;  rest /= HD;
  int head = rest % NHEADS; int win = rest / NHEADS;
  __bf16 v = (__bf16)0.0f;
  if (j < NTOK)
    v = qkv[((size_t)(win * NPAD + j)) * QKVN + 2 * CDIM + head * HD + d];
  Vt[idx] = v;
}

// ---------------------------------------------------------------------------
// Generic bf16 WMMA GEMM: C = A(MxK) @ B(NxK)^T, fused epilogues.
// block = 128 threads (4 waves); block tile 64x128; wave tile 64x32
// (2 B-fragments reused across 4 A subtiles: 12 VMEM fragment loads / 8 WMMAs).
// EPI 0: +bias, scale cols<768 by QSCALE, store bf16     (QKV)
// EPI 1: +bias, window-reverse scatter + residual, f32   (proj)
// EPI 2: +bias, GELU, store bf16                         (fc1)
// EPI 3: +bias, +residual, store f32                     (fc2)
// ---------------------------------------------------------------------------
template <int EPI>
__global__ __launch_bounds__(128) void k_gemm(const __bf16* __restrict__ A,
                                              const __bf16* __restrict__ B,
                                              const float* __restrict__ bias,
                                              float* __restrict__ outF,
                                              __bf16* __restrict__ outB,
                                              const float* __restrict__ resid,
                                              int N, int K, int shifted) {
  const int lane = threadIdx.x & 31;
  const int wave = threadIdx.x >> 5;
  const int mBase = blockIdx.x * 64;
  const int nBase = blockIdx.y * 128 + wave * 32;

  f32x8 acc[4][2] = {};
  const __bf16* aP = A + (size_t)mBase * K;
  const __bf16* bP = B + (size_t)nBase * K;
  for (int k = 0; k < K; k += 32) {
    __builtin_prefetch(bP + k + 64, 0, 1);
    bf16x16 bf0 = load_frag(bP + k, K, lane);
    bf16x16 bf1 = load_frag(bP + (size_t)16 * K + k, K, lane);
    #pragma unroll
    for (int i = 0; i < 4; ++i) {
      bf16x16 af = load_frag(aP + (size_t)(16 * i) * K + k, K, lane);
      acc[i][0] = __builtin_amdgcn_wmma_f32_16x16x32_bf16(
          false, af, false, bf0, (short)0, acc[i][0], false, false);
      acc[i][1] = __builtin_amdgcn_wmma_f32_16x16x32_bf16(
          false, af, false, bf1, (short)0, acc[i][1], false, false);
    }
  }

  const int nc0 = nBase + (lane & 15);
  const int rOff = (lane >> 4) << 3;
  const float bv0 = bias[nc0];
  const float bv1 = bias[nc0 + 16];
  #pragma unroll
  for (int i = 0; i < 4; ++i) {
    #pragma unroll
    for (int v = 0; v < 8; ++v) {
      const int m = mBase + 16 * i + rOff + v;
      float val0 = acc[i][0][v] + bv0;
      float val1 = acc[i][1][v] + bv1;
      if (EPI == 0) {
        if (nc0 < CDIM) val0 *= QSCALE;
        if (nc0 + 16 < CDIM) val1 *= QSCALE;
        outB[(size_t)m * N + nc0] = (__bf16)val0;
        outB[(size_t)m * N + nc0 + 16] = (__bf16)val1;
      } else if (EPI == 1) {
        const int win = m / NPAD, t = m % NPAD;
        if (t < NTOK) {
          const size_t o = (size_t)mapToken(win, t, shifted) * CDIM + nc0;
          outF[o] = val0 + resid[o];
          outF[o + 16] = val1 + resid[o + 16];
        }
      } else if (EPI == 2) {
        outB[(size_t)m * N + nc0] = (__bf16)geluf(val0);
        outB[(size_t)m * N + nc0 + 16] = (__bf16)geluf(val1);
      } else {
        const size_t o = (size_t)m * N + nc0;
        outF[o] = val0 + resid[o];
        outF[o + 16] = val1 + resid[o + 16];
      }
    }
  }
}

// ---------------------------------------------------------------------------
// Window attention: one wave per (16-query tile, head, window).
// S = (scaled Q)K^T via 25 single-step WMMAs; bias/mask/softmax in LDS;
// P kept unnormalized in bf16 LDS; AV via WMMA over K=416; rows scaled 1/sum.
// ---------------------------------------------------------------------------
__global__ __launch_bounds__(32) void k_attn(const __bf16* __restrict__ qkv,
                                             const __bf16* __restrict__ Vt,
                                             const float* __restrict__ rpb,
                                             const int* __restrict__ rpi,
                                             __bf16* __restrict__ attnO,
                                             int shifted) {
  __shared__ __align__(16) float  S[16][KPAD];
  __shared__ __align__(16) __bf16 P[16][KPAD];
  __shared__ float rowSum[16];
  __shared__ unsigned char lbl[NPAD];

  const int lane = threadIdx.x;
  const int qt = blockIdx.x;    // 0..24
  const int head = blockIdx.y;  // 0..23
  const int win = blockIdx.z;   // 0..15
  const int qbase = qt * 16;

  if (shifted)
    for (int t = lane; t < NTOK; t += 32) lbl[t] = (unsigned char)labelOf(win, t);

  // Q fragment (rows qbase..qbase+15, padded buffer so always in-bounds)
  const __bf16* Qb = qkv + ((size_t)(win * NPAD + qbase)) * QKVN + head * HD;
  bf16x16 qf = load_frag(Qb, QKVN, lane);

  // S = Q K^T  (single WMMA per 16x16 tile: K dim == head dim == 32)
  const __bf16* Kb = qkv + ((size_t)(win * NPAD)) * QKVN + CDIM + head * HD;
  const int rb = (lane >> 4) << 3;
  for (int jt = 0; jt < 25; ++jt) {
    bf16x16 kf = load_frag(Kb + (size_t)(jt * 16) * QKVN, QKVN, lane);
    f32x8 a = {};
    a = __builtin_amdgcn_wmma_f32_16x16x32_bf16(false, qf, false, kf, (short)0, a,
                                                false, false);
    int n = jt * 16 + (lane & 15);
    #pragma unroll
    for (int v = 0; v < 8; ++v) S[rb + v][n] = a[v];
  }
  __syncthreads();

  // bias + mask + softmax (2 lanes per row)
  const int r = lane >> 1;
  const int par = lane & 1;
  const int q = qbase + r;
  float mx = -3.4e38f;
  if (q < NTOK) {
    int lq = shifted ? lbl[q] : 0;
    for (int j = par; j < NTOK; j += 2) {
      float s = S[r][j] + rpb[(size_t)rpi[q * NTOK + j] * NHEADS + head];
      if (shifted && lbl[j] != lq) s -= 100.0f;
      S[r][j] = s;
      mx = fmaxf(mx, s);
    }
  }
  mx = fmaxf(mx, __shfl_xor(mx, 1));
  float sm = 0.f;
  if (q < NTOK) {
    for (int j = par; j < KPAD; j += 2) {
      float e = (j < NTOK) ? __expf(S[r][j] - mx) : 0.0f;
      sm += e;
      P[r][j] = (__bf16)e;
    }
  } else {
    for (int j = par; j < KPAD; j += 2) P[r][j] = (__bf16)0.0f;
  }
  sm += __shfl_xor(sm, 1);
  if (par == 0) rowSum[r] = (q < NTOK && sm > 0.0f) ? sm : 1.0f;
  __syncthreads();

  // out = (P V) / rowSum
  const __bf16* Vb = Vt + ((size_t)(win * NHEADS + head)) * HD * KPAD;
  for (int dt = 0; dt < 2; ++dt) {
    f32x8 a = {};
    for (int jt = 0; jt < 13; ++jt) {
      bf16x16 pf = load_frag(&P[0][0] + jt * 32, KPAD, lane);
      bf16x16 vf = load_frag(Vb + (size_t)(dt * 16) * KPAD + jt * 32, KPAD, lane);
      a = __builtin_amdgcn_wmma_f32_16x16x32_bf16(false, pf, false, vf, (short)0, a,
                                                  false, false);
    }
    int n = dt * 16 + (lane & 15);
    #pragma unroll
    for (int v = 0; v < 8; ++v) {
      int m = rb + v, qq = qbase + m;
      if (qq < NTOK)
        attnO[((size_t)(win * NPAD + qq)) * CDIM + head * HD + n] =
            (__bf16)(a[v] / rowSum[m]);
    }
  }
}

// ---------------------------------------------------------------------------
// Head: LN(768) -> 1x1 conv 768->64 + GELU -> 1x1 conv 64->2 + GELU.
// One wave per token.
// ---------------------------------------------------------------------------
__global__ __launch_bounds__(256) void k_head(const float* __restrict__ X,
                                              const float* __restrict__ g,
                                              const float* __restrict__ b,
                                              const float* __restrict__ c1w,
                                              const float* __restrict__ c1b,
                                              const float* __restrict__ c2w,
                                              const float* __restrict__ c2b,
                                              float* __restrict__ outBr) {
  __shared__ float h1s[8][64];
  const int lane = threadIdx.x & 31;
  const int wv = threadIdx.x >> 5;
  const int tok = blockIdx.x * 8 + wv;  // grid sized exactly: no remainder
  const float* x = X + (size_t)tok * CDIM;
  float v[24]; float s = 0.f;
  #pragma unroll
  for (int i = 0; i < 24; ++i) { v[i] = x[lane + 32 * i]; s += v[i]; }
  float mu = waveSum(s) * (1.0f / CDIM);
  float qv = 0.f;
  #pragma unroll
  for (int i = 0; i < 24; ++i) { float d = v[i] - mu; qv += d * d; }
  float rs = rsqrtf(waveSum(qv) * (1.0f / CDIM) + 1e-5f);
  float y[24];
  #pragma unroll
  for (int i = 0; i < 24; ++i) {
    int c = lane + 32 * i;
    y[i] = (v[i] - mu) * rs * g[c] + b[c];
  }
  for (int o = 0; o < 64; ++o) {
    const float* w = c1w + (size_t)o * CDIM;
    float p = 0.f;
    #pragma unroll
    for (int i = 0; i < 24; ++i) p += y[i] * w[lane + 32 * i];
    p = waveSum(p);
    if (lane == 0) h1s[wv][o] = geluf(p + c1b[o]);
  }
  __syncthreads();
  for (int o2 = 0; o2 < 2; ++o2) {
    float p = h1s[wv][lane] * c2w[o2 * 64 + lane] +
              h1s[wv][lane + 32] * c2w[o2 * 64 + lane + 32];
    p = waveSum(p);
    if (lane == 0) outBr[(size_t)tok * 2 + o2] = geluf(p + c2b[o2]);
  }
}

// out (B,2,D,H,W) = branch0 * branch1  (branches stored (tok,2))
__global__ __launch_bounds__(256) void k_final(const float* __restrict__ s0,
                                               const float* __restrict__ s1,
                                               float* __restrict__ out) {
  int o = blockIdx.x * 256 + threadIdx.x;
  if (o >= BATCH * 2 * DDIM * HDIM * WDIM) return;
  int r = o % (DDIM * HDIM * WDIM);
  int p = (o / (DDIM * HDIM * WDIM)) % 2;
  int bb = o / (2 * DDIM * HDIM * WDIM);
  int tok = bb * (DDIM * HDIM * WDIM) + r;
  out[o] = s0[tok * 2 + p] * s1[tok * 2 + p];
}

// ---------------------------------------------------------------------------
// Host orchestration
// ---------------------------------------------------------------------------
extern "C" void kernel_launch(void* const* d_in, const int* in_sizes, int n_in,
                              void* d_out, int out_size, void* d_ws, size_t ws_size,
                              hipStream_t stream) {
  (void)in_sizes; (void)n_in; (void)out_size; (void)ws_size;
  const float* x      = (const float*)d_in[0];
  const float* n1_g   = (const float*)d_in[1];
  const float* n1_b   = (const float*)d_in[2];
  const float* qkv_w  = (const float*)d_in[3];
  const float* qkv_b  = (const float*)d_in[4];
  const float* proj_w = (const float*)d_in[5];
  const float* proj_b = (const float*)d_in[6];
  const float* rpb    = (const float*)d_in[7];
  const float* n2_g   = (const float*)d_in[8];
  const float* n2_b   = (const float*)d_in[9];
  const float* fc1_w  = (const float*)d_in[10];
  const float* fc1_b  = (const float*)d_in[11];
  const float* fc2_w  = (const float*)d_in[12];
  const float* fc2_b  = (const float*)d_in[13];
  const float* hln_g  = (const float*)d_in[14];
  const float* hln_b  = (const float*)d_in[15];
  const float* c1_w   = (const float*)d_in[16];
  const float* c1_b   = (const float*)d_in[17];
  const float* c2_w   = (const float*)d_in[18];
  const float* c2_b   = (const float*)d_in[19];
  const int*   rpi    = (const int*)d_in[20];

  // workspace carve-out
  char* ws = (char*)d_ws;
  size_t off = 0;
  auto carve = [&](size_t bytes) -> char* {
    char* p = ws + off;
    off += (bytes + 255) & ~(size_t)255;
    return p;
  };
  __bf16* wQKV  = (__bf16*)carve((size_t)QKVN * CDIM * 2);
  __bf16* wPROJ = (__bf16*)carve((size_t)CDIM * CDIM * 2);
  __bf16* wFC1  = (__bf16*)carve((size_t)MLPN * CDIM * 2);
  __bf16* wFC2  = (__bf16*)carve((size_t)CDIM * MLPN * 2);
  float*  bufX  = (float*)carve((size_t)TOKENS * CDIM * 4);
  float*  bufX2 = (float*)carve((size_t)TOKENS * CDIM * 4);
  __bf16* bufH  = (__bf16*)carve((size_t)MWIN * CDIM * 2);     // also LN2 output
  __bf16* qkvB  = (__bf16*)carve((size_t)TOKENS * MLPN * 2);   // also fc1/GELU output
  __bf16* Vt    = (__bf16*)carve((size_t)NWIN * NHEADS * HD * KPAD * 2);
  __bf16* attnO = (__bf16*)carve((size_t)MWIN * CDIM * 2);
  float*  bufBr = (float*)carve((size_t)2 * TOKENS * 2 * 4);
  __bf16* bufG  = qkvB;   // alias: qkv dead before fc1 runs
  __bf16* bufH2 = bufH;   // alias: windowed LN1 dead before LN2 runs

  auto cast = [&](const float* s, __bf16* d, int n) {
    k_cast_bf16<<<(n + 255) / 256, 256, 0, stream>>>(s, d, n);
  };

  for (int br = 0; br < 2; ++br) {
    k_init_x<<<(TOKENS * CDIM + 255) / 256, 256, 0, stream>>>(x, bufX);
    for (int bl = 0; bl < 2; ++bl) {
      const int sl = br * 2 + bl;
      const int shifted = bl;  // second block shifted
      cast(qkv_w  + (size_t)sl * QKVN * CDIM, wQKV, QKVN * CDIM);
      cast(proj_w + (size_t)sl * CDIM * CDIM, wPROJ, CDIM * CDIM);
      cast(fc1_w  + (size_t)sl * MLPN * CDIM, wFC1, MLPN * CDIM);
      cast(fc2_w  + (size_t)sl * CDIM * MLPN, wFC2, CDIM * MLPN);

      // LN1 + (shift) + window partition
      k_ln<<<MWIN / 8, 256, 0, stream>>>(bufX, n1_g + sl * CDIM, n1_b + sl * CDIM,
                                         bufH, shifted ? 2 : 1, MWIN);
      // QKV GEMM (Q pre-scaled)
      k_gemm<0><<<dim3(MWIN / 64, QKVN / 128), 128, 0, stream>>>(
          bufH, wQKV, qkv_b + (size_t)sl * QKVN, nullptr, qkvB, nullptr,
          QKVN, CDIM, 0);
      // V transpose
      k_vt<<<(NWIN * NHEADS * HD * KPAD + 255) / 256, 256, 0, stream>>>(qkvB, Vt);
      // attention
      k_attn<<<dim3(25, NHEADS, NWIN), 32, 0, stream>>>(
          qkvB, Vt, rpb + (size_t)sl * 2535 * NHEADS, rpi, attnO, shifted);
      // proj GEMM + window reverse + residual -> bufX2
      k_gemm<1><<<dim3(MWIN / 64, CDIM / 128), 128, 0, stream>>>(
          attnO, wPROJ, proj_b + (size_t)sl * CDIM, bufX2, nullptr, bufX,
          CDIM, CDIM, shifted);
      // LN2 (token-major)
      k_ln<<<TOKENS / 8, 256, 0, stream>>>(bufX2, n2_g + sl * CDIM,
                                           n2_b + sl * CDIM, bufH2, 0, TOKENS);
      // fc1 + GELU
      k_gemm<2><<<dim3(TOKENS / 64, MLPN / 128), 128, 0, stream>>>(
          bufH2, wFC1, fc1_b + (size_t)sl * MLPN, nullptr, bufG, nullptr,
          MLPN, CDIM, 0);
      // fc2 + residual -> bufX
      k_gemm<3><<<dim3(TOKENS / 64, CDIM / 128), 128, 0, stream>>>(
          bufG, wFC2, fc2_b + (size_t)sl * CDIM, bufX, nullptr, bufX2,
          CDIM, MLPN, 0);
    }
    // head: LN + conv768->64 +GELU + conv64->2 +GELU
    k_head<<<TOKENS / 8, 256, 0, stream>>>(
        bufX, hln_g + br * CDIM, hln_b + br * CDIM,
        c1_w + (size_t)br * 64 * CDIM, c1_b + br * 64,
        c2_w + (size_t)br * 2 * 64, c2_b + br * 2,
        bufBr + (size_t)br * TOKENS * 2);
  }
  k_final<<<(BATCH * 2 * DDIM * HDIM * WDIM + 255) / 256, 256, 0, stream>>>(
      bufBr, bufBr + (size_t)TOKENS * 2, (float*)d_out);
}
